// FFSNN_64862596104551
// MI455X (gfx1250) — matrix operand order repaired and checked by reading, not compiled, not run on hardware
//
#include <hip/hip_runtime.h>

// ---------------------------------------------------------------------------
// CDNA5 (gfx1250) feedforward SNN: per-timestep [GEMM -> Leaky LIF] x 2.
//  - V_WMMA_F32_16X16X32_F16 with f32 accumulation
//  - weights pre-converted to f16 once per launch (reused 64 steps, L2-resident)
//  - Tensor Data Mover (TENSOR_LOAD_TO_LDS) stages f16 tiles into padded LDS
//  - double-buffered LDS, TDM overlapped with WMMA via TENSORcnt pipelining
// Requires ws_size >= 13 MB (W1h 4MB + W2h 8MB + spk1 f16 1MB).
// ---------------------------------------------------------------------------

typedef __attribute__((ext_vector_type(16))) _Float16     v16h;
typedef __attribute__((ext_vector_type(8)))  float        v8f;
typedef __attribute__((ext_vector_type(4)))  unsigned int u32x4;
typedef __attribute__((ext_vector_type(8)))  int          i32x8;
typedef __attribute__((ext_vector_type(4)))  int          i32x4;

#define SNN_B    256
#define SNN_T    64
#define SNN_DIN  1024
#define SNN_H    2048

#define BM 64            // macro tile rows (batch)
#define BN 128           // macro tile cols (features)
#define BK 32            // K chunk == WMMA K
#define LDS_STRIDE 20    // dwords per row: 16 data + 4 pad (80B, 16B aligned)

#if __has_builtin(__builtin_amdgcn_tensor_load_to_lds) && __has_builtin(__builtin_amdgcn_s_wait_tensorcnt)
#define SNN_USE_TDM 1
#else
#define SNN_USE_TDM 0
#endif

__device__ __forceinline__ unsigned int pack_f16x2(float a, float b) {
    _Float16 lo = (_Float16)a;
    _Float16 hi = (_Float16)b;
    unsigned int ul = (unsigned int)__builtin_bit_cast(unsigned short, lo);
    unsigned int uh = (unsigned int)__builtin_bit_cast(unsigned short, hi);
    return ul | (uh << 16);
}

// --- TDM descriptor helpers (CDNA5 ISA) -------------
// Group0: [1:0]=count=1, [63:32]=lds_addr, [120:64]=global_addr, [127:126]=type=2
__device__ __forceinline__ u32x4 tdm_g0(unsigned lds_addr, unsigned long long gaddr) {
    u32x4 g0;
    g0.x = 1u;
    g0.y = lds_addr;
    g0.z = (unsigned int)gaddr;
    g0.w = (unsigned int)((gaddr >> 32) & 0x01FFFFFFull) | (2u << 30);
    return g0;
}
// Group1: data_size=1(2B), pad_enable=1, pad_interval=3(16dw), pad_amount=3(4dw),
// huge tensor dims (no OOB), tile BKxRows, dim0 stride = K elements.
__device__ __forceinline__ i32x8 tdm_g1(int tile_rows, int k_stride) {
    const unsigned td0 = 0x7FFFFFFFu, td1 = 0x7FFFFFFFu;
    i32x8 g1;
    g1[0] = (int)((1u << 16) | (1u << 20) | (3u << 22) | (3u << 25));
    g1[1] = (int)((td0 & 0xFFFFu) << 16);                    // [15:0] atomic_barrier=0
    g1[2] = (int)((td0 >> 16) | ((td1 & 0xFFFFu) << 16));
    g1[3] = (int)((td1 >> 16) | ((unsigned)BK << 16));       // tile_dim0 = BK
    g1[4] = tile_rows;                                       // tile_dim1; tile_dim2=0
    g1[5] = k_stride;                                        // tensor_dim0_stride lo
    g1[6] = 0;
    g1[7] = 0;
    return g1;
}
__device__ __forceinline__ void tdm_issue(u32x4 g0, i32x8 g1) {
#if SNN_USE_TDM
    const i32x4 z4 = {0, 0, 0, 0};
    const i32x8 z8 = {0, 0, 0, 0, 0, 0, 0, 0};
    // amdgpu-toolchain (clang-23) 6-arg form: (g0, g1, g2, g3, g4, cpol)
    __builtin_amdgcn_tensor_load_to_lds(g0, g1, z4, z4, z8, 0);
#else
    (void)g0; (void)g1;
#endif
}
template <int N>
__device__ __forceinline__ void tdm_wait() {
#if SNN_USE_TDM
    __builtin_amdgcn_s_wait_tensorcnt((short)N);
#endif
}

// ---------------------------------------------------------------------------
// one-time f32 -> packed f16 conversion (weights), 4 elems / thread
__global__ __launch_bounds__(256)
void cvt_f32_f16_kernel(const float* __restrict__ src, unsigned int* __restrict__ dst) {
    const size_t i = (size_t)blockIdx.x * 256 + threadIdx.x;
    const float4 v = reinterpret_cast<const float4*>(src)[i];
    uint2 o;
    o.x = pack_f16x2(v.x, v.y);
    o.y = pack_f16x2(v.z, v.w);
    reinterpret_cast<uint2*>(dst)[i] = o;
}

// ---------------------------------------------------------------------------
// A_F16 : A operand is f16 (spike matrix) -> staged by TDM
// SPK16 : this layer's spike state stored as f16 (layer1) vs f32 (layer2)
template <bool A_F16, bool SPK16>
__global__ __launch_bounds__(256)
void snn_layer_kernel(const void* __restrict__ Asrc, int a_stride,
                      const void* __restrict__ Wh,   // f16 weights [H,K]
                      int K,
                      const float* __restrict__ bias,
                      float* __restrict__ mem,
                      void* __restrict__ spk,        // prev in / new out (same buffer)
                      int is_first)
{
    __shared__ __align__(16) unsigned int As[2][BM][LDS_STRIDE];
    __shared__ __align__(16) unsigned int Bs[2][BN][LDS_STRIDE];

    const int tid  = threadIdx.x;
    const int lane = tid & 31;
    const int wave = tid >> 5;
    const int hf   = lane >> 4;
    const int l16  = lane & 15;
    const int m_off = (wave >> 2) * 32;
    const int n_off = (wave & 3)  * 32;
    const int m0 = blockIdx.x * BM;
    const int n0 = blockIdx.y * BN;
    const int nchunks = K / BK;

    // uniform byte addresses for TDM
    const unsigned long long WhA = (unsigned long long)(uintptr_t)Wh +
                                   2ull * (unsigned long long)n0 * (unsigned)K;
    const unsigned long long AhA = (unsigned long long)(uintptr_t)Asrc +
                                   2ull * (unsigned long long)m0 * (unsigned)a_stride;
    const unsigned ldsB[2] = { (unsigned)(uintptr_t)&Bs[0][0][0],
                               (unsigned)(uintptr_t)&Bs[1][0][0] };
    const unsigned ldsA[2] = { (unsigned)(uintptr_t)&As[0][0][0],
                               (unsigned)(uintptr_t)&As[1][0][0] };
    const i32x8 g1B = tdm_g1(BN, K);
    const i32x8 g1A = tdm_g1(BM, a_stride);

    // stage chunk c into buffer c&1 (TDM for f16 tiles, VALU-cvt for f32 A)
    auto issue_chunk = [&](int c) {
        const int buf = c & 1;
        const int kc  = c * BK;
        if constexpr (SNN_USE_TDM) {
            if (tid < 32) {
                tdm_issue(tdm_g0(ldsB[buf], WhA + 2ull * (unsigned)kc), g1B);
                if constexpr (A_F16)
                    tdm_issue(tdm_g0(ldsA[buf], AhA + 2ull * (unsigned)kc), g1A);
            }
        } else {
            const _Float16* Wp = (const _Float16*)Wh;
#pragma unroll
            for (int i = 0; i < 2; ++i) {
                int idx = tid + 256 * i;          // 512 uint4
                int r = idx >> 2;                 // 0..127
                int q = idx & 3;
                const uint4 v = *reinterpret_cast<const uint4*>(
                    Wp + (size_t)(n0 + r) * K + kc + q * 8);
                *reinterpret_cast<uint4*>(&Bs[buf][r][4 * q]) = v;
            }
            if constexpr (A_F16) {
                const _Float16* Ap = (const _Float16*)Asrc;
                int r = tid >> 2;                 // 0..63
                int q = tid & 3;
                const uint4 v = *reinterpret_cast<const uint4*>(
                    Ap + (size_t)(m0 + r) * a_stride + kc + q * 8);
                *reinterpret_cast<uint4*>(&As[buf][r][4 * q]) = v;
            }
        }
        if constexpr (!A_F16) {                    // layer1: x is f32, cvt-pack
            const float* A = (const float*)Asrc;
#pragma unroll
            for (int i = 0; i < 2; ++i) {
                int idx = tid + 256 * i;          // 512 float4
                int r = idx >> 3;                 // 0..63
                int q = idx & 7;
                const float4 v = *reinterpret_cast<const float4*>(
                    A + (size_t)(m0 + r) * a_stride + kc + q * 4);
                As[buf][r][2 * q]     = pack_f16x2(v.x, v.y);
                As[buf][r][2 * q + 1] = pack_f16x2(v.z, v.w);
            }
        }
    };

    const v8f vzero = {};
    v8f acc[2][2];
#pragma unroll
    for (int mt = 0; mt < 2; ++mt)
#pragma unroll
        for (int nt = 0; nt < 2; ++nt)
            acc[mt][nt] = vzero;

    issue_chunk(0);

    for (int c = 0; c < nchunks; ++c) {
        const int buf = c & 1;
        const bool more = (c + 1 < nchunks);

        __syncthreads();                 // all readers of buf^1 (chunk c-1) done
        if (more) issue_chunk(c + 1);    // DMA next chunk into the other buffer
        if constexpr (SNN_USE_TDM) {
            if (tid < 32) {              // in-order TDM: allow chunk c+1 in flight
                if (more) tdm_wait<A_F16 ? 2 : 1>();
                else      tdm_wait<0>();
            }
        }
        __syncthreads();                 // chunk c resident in LDS for all waves

        // ---- fragments (contiguous k: element s of half h is k = 16h + s, A and B alike) + WMMA ----
        v16h afrag[2], bfrag[2];
#pragma unroll
        for (int mt = 0; mt < 2; ++mt) {
            const unsigned int* row = As[buf][m_off + mt * 16 + l16];
            uint4* ap = reinterpret_cast<uint4*>(&afrag[mt]);
            ap[0] = *reinterpret_cast<const uint4*>(row + 8 * hf);
            ap[1] = *reinterpret_cast<const uint4*>(row + 8 * hf + 4);
        }
#pragma unroll
        for (int nt = 0; nt < 2; ++nt) {
            const unsigned int* row = Bs[buf][n_off + nt * 16 + l16];
            uint4* bp = reinterpret_cast<uint4*>(&bfrag[nt]);
            bp[0] = *reinterpret_cast<const uint4*>(row + 8 * hf);
            bp[1] = *reinterpret_cast<const uint4*>(row + 8 * hf + 4);
        }
#pragma unroll
        for (int mt = 0; mt < 2; ++mt)
#pragma unroll
            for (int nt = 0; nt < 2; ++nt)
                acc[mt][nt] = __builtin_amdgcn_wmma_f32_16x16x32_f16(
                    false, afrag[mt], false, bfrag[nt],
                    (short)0, acc[mt][nt], false, false);
    }

    // ---- epilogue: bias + Leaky LIF (reset-by-subtract, heaviside) ----
#pragma unroll
    for (int mt = 0; mt < 2; ++mt) {
#pragma unroll
        for (int nt = 0; nt < 2; ++nt) {
            const int col = n0 + n_off + nt * 16 + l16;
            const float bcol = bias[col];
#pragma unroll
            for (int r = 0; r < 8; ++r) {
                const int m = m0 + m_off + mt * 16 + 8 * hf + r;  // M = r + 8*half
                const size_t idx = (size_t)m * SNN_H + col;
                const float cur = acc[mt][nt][r] + bcol;
                float memp = 0.0f, spkp = 0.0f;
                if (!is_first) {
                    memp = mem[idx];
                    if constexpr (SPK16)
                        spkp = (float)((const _Float16*)spk)[idx];
                    else
                        spkp = ((const float*)spk)[idx];
                }
                const float mv = 0.9f * memp + cur - spkp;  // beta*mem + I - thr*spk
                mem[idx] = mv;
                const float sv = (mv > 1.0f) ? 1.0f : 0.0f; // heaviside(mem - thr)
                if constexpr (SPK16)
                    ((_Float16*)spk)[idx] = (_Float16)sv;
                else
                    ((float*)spk)[idx] = sv;
            }
        }
    }
}

extern "C" void kernel_launch(void* const* d_in, const int* in_sizes, int n_in,
                              void* d_out, int out_size, void* d_ws, size_t ws_size,
                              hipStream_t stream) {
    (void)in_sizes; (void)n_in; (void)out_size; (void)ws_size;

    const float* x  = (const float*)d_in[0];   // [B, T, DIN]
    const float* W1 = (const float*)d_in[1];   // [H, DIN]
    const float* b1 = (const float*)d_in[2];   // [H]
    const float* W2 = (const float*)d_in[3];   // [H, H]
    const float* b2 = (const float*)d_in[4];   // [H]

    // d_out = concat(spk2[B,H], mem1[B,H], mem2[B,H]); mem carries live in-place.
    float* spk2 = (float*)d_out;
    float* mem1 = (float*)d_out + (size_t)SNN_B * SNN_H;
    float* mem2 = (float*)d_out + 2 * (size_t)SNN_B * SNN_H;

    // workspace: W1h (4MB) | W2h (8MB) | spk1 f16 (1MB)
    char* ws = (char*)d_ws;
    void* W1h  = (void*)ws;
    void* W2h  = (void*)(ws + (size_t)SNN_H * SNN_DIN * 2);
    void* spk1 = (void*)(ws + (size_t)SNN_H * SNN_DIN * 2 + (size_t)SNN_H * SNN_H * 2);

    // one-time weight conversion (runs every launch; deterministic)
    cvt_f32_f16_kernel<<<(SNN_H * SNN_DIN / 4) / 256, 256, 0, stream>>>(
        W1, (unsigned int*)W1h);
    cvt_f32_f16_kernel<<<((size_t)SNN_H * SNN_H / 4) / 256, 256, 0, stream>>>(
        W2, (unsigned int*)W2h);

    const dim3 grid(SNN_B / BM, SNN_H / BN);   // (4, 16)
    const dim3 block(256);

    for (int t = 0; t < SNN_T; ++t) {
        // layer 1: cur1 = x_t @ W1^T + b1 ; LIF -> mem1 (f32), spk1 (f16 ws)
        snn_layer_kernel<false, true><<<grid, block, 0, stream>>>(
            (const void*)(x + (size_t)t * SNN_DIN), SNN_T * SNN_DIN,
            W1h, SNN_DIN, b1, mem1, spk1, t == 0 ? 1 : 0);
        // layer 2: cur2 = spk1 @ W2^T + b2 ; LIF -> mem2 (f32), spk2 (f32 d_out)
        snn_layer_kernel<true, false><<<grid, block, 0, stream>>>(
            spk1, SNN_H,
            W2h, SNN_H, b2, mem2, (void*)spk2, t == 0 ? 1 : 0);
    }
}
